// GATRegressor_52312701665966
// MI455X (gfx1250) — matrix operand order repaired and checked by reading, not compiled, not run on hardware
//
#include <hip/hip_runtime.h>
#include <hip/hip_bf16.h>

typedef __attribute__((ext_vector_type(16))) _Float16 v16h;
typedef __attribute__((ext_vector_type(8)))  float    v8f;

#define NEG_SLOPE 0.2f

// ---------------- order-preserving float<->uint encoding for atomicMax ----
__device__ __forceinline__ unsigned enc_f32(float f) {
    unsigned u = __float_as_uint(f);
    return (u & 0x80000000u) ? ~u : (u | 0x80000000u);
}
__device__ __forceinline__ float dec_f32(unsigned u) {
    unsigned v = (u & 0x80000000u) ? (u & 0x7FFFFFFFu) : ~u;
    return __uint_as_float(v);
}
// encoded(-inf) = ~0xFF800000 = 0x007FFFFF
#define ENC_NEG_INF 0x007FFFFFu

// ---------------- WMMA GEMM:  C[nrows x NCOLS] = A[nrows x K] * W[K x NCOLS]
// f32 inputs converted to f16, f32 accumulate via v_wmma_f32_16x16x32_f16.
// Block = 256 threads = 8 waves; each wave owns a 16-row stripe (128 rows/block).
// W is staged transposed in LDS as f16 so each lane's B-fragment is a
// contiguous 32B run (2x ds_load_b128).
template <int K, int NCOLS>
__global__ void wmma_gemm_kernel(const float* __restrict__ A,
                                 const float* __restrict__ W,
                                 float* __restrict__ C,
                                 int nrows) {
    constexpr int LDSK = K + 16;                      // halves; keeps 32B align (stride 288B)
    __shared__ alignas(32) _Float16 Wt[NCOLS * LDSK];

    // cooperative load + f32->f16 convert, transposed: Wt[n][k]
    for (int idx = threadIdx.x; idx < K * NCOLS; idx += blockDim.x) {
        int k = idx / NCOLS;
        int n = idx - k * NCOLS;
        Wt[n * LDSK + k] = (_Float16)W[idx];
    }
    __syncthreads();

    const int wave = threadIdx.x >> 5;
    const int lane = threadIdx.x & 31;
    const int half = lane >> 4;        // 0: lanes 0-15, 1: lanes 16-31
    const int l16  = lane & 15;
    const int m0   = blockIdx.x * 128 + wave * 16;
    const int row  = m0 + l16;
    const bool rowok = (row < nrows);

    v8f acc[NCOLS / 16];
#pragma unroll
    for (int t = 0; t < NCOLS / 16; ++t)
#pragma unroll
        for (int i = 0; i < 8; ++i) acc[t][i] = 0.0f;

#pragma unroll
    for (int kk = 0; kk < K; kk += 32) {
        // ---- A fragment (contiguous k order, matches B):
        // lane<16 : K = kk+0..15
        // lane>=16: K = kk+16..31
        v16h a;
        if (rowok) {
            const float* ap = A + (size_t)row * K + kk + half * 16;
#pragma unroll
            for (int i = 0; i < 8; ++i) a[i] = (_Float16)ap[i];
#pragma unroll
            for (int i = 0; i < 8; ++i) a[8 + i] = (_Float16)ap[8 + i];
        } else {
#pragma unroll
            for (int i = 0; i < 16; ++i) a[i] = (_Float16)0.0f;
        }
#pragma unroll
        for (int t = 0; t < NCOLS / 16; ++t) {
            // ---- B fragment: lane<16 -> Wt[n][kk..kk+15], lane>=16 -> Wt[n][kk+16..kk+31]
            const _Float16* bp = &Wt[(t * 16 + l16) * LDSK + kk + half * 16];
            v16h b = *(const v16h*)bp;
            acc[t] = __builtin_amdgcn_wmma_f32_16x16x32_f16(
                false, a, false, b, (short)0, acc[t], false, false);
        }
    }

    // ---- store: C/D layout -> VGPR r holds (M = r + 8*half, N = l16)
#pragma unroll
    for (int t = 0; t < NCOLS / 16; ++t) {
#pragma unroll
        for (int r = 0; r < 8; ++r) {
            int orow = m0 + r + half * 8;
            if (orow < nrows)
                C[(size_t)orow * NCOLS + t * 16 + l16] = acc[t][r];
        }
    }
}

// ---------------- per-node attention coefficients: a = sum_d h[n,h,d]*att[h,d]
// one wave per node; F = heads*64, head = c>>6
__global__ void att_coef_kernel(const float* __restrict__ H,
                                const float* __restrict__ att_src,
                                const float* __restrict__ att_dst,
                                float* __restrict__ a_src,
                                float* __restrict__ a_dst,
                                int n, int heads) {
    const int wave = threadIdx.x >> 5;
    const int lane = threadIdx.x & 31;
    const int node = blockIdx.x * (blockDim.x >> 5) + wave;
    if (node >= n) return;
    const int F = heads * 64;
    float ss[2] = {0.f, 0.f}, sd[2] = {0.f, 0.f};
    for (int c = lane; c < F; c += 32) {
        float h = H[(size_t)node * F + c];
        int hd = c >> 6;
        ss[hd] += h * att_src[c];
        sd[hd] += h * att_dst[c];
    }
#pragma unroll
    for (int m = 16; m >= 1; m >>= 1) {
        ss[0] += __shfl_xor(ss[0], m, 32);
        ss[1] += __shfl_xor(ss[1], m, 32);
        sd[0] += __shfl_xor(sd[0], m, 32);
        sd[1] += __shfl_xor(sd[1], m, 32);
    }
    if (lane == 0) {
        for (int hh = 0; hh < heads; ++hh) {
            a_src[(size_t)node * heads + hh] = ss[hh];
            a_dst[(size_t)node * heads + hh] = sd[hh];
        }
    }
}

__global__ void init_softmax_kernel(unsigned* __restrict__ m,
                                    float* __restrict__ denom, int n) {
    int i = blockIdx.x * blockDim.x + threadIdx.x;
    if (i < n) { m[i] = ENC_NEG_INF; denom[i] = 0.0f; }
}

// ---------------- edge pass 1: e = leakyrelu(a_src[s]+a_dst[d]); segment max
template <int H>
__global__ void edge_max_kernel(const long long* __restrict__ srcI,
                                const long long* __restrict__ dstI,
                                const float* __restrict__ a_src,
                                const float* __restrict__ a_dst,
                                float* __restrict__ ebuf,
                                unsigned* __restrict__ mmax,
                                int Ereal, int nnodes) {
    int e = blockIdx.x * blockDim.x + threadIdx.x;
    if (e >= Ereal + nnodes) return;
    int s, d;
    if (e < Ereal) { s = (int)srcI[e]; d = (int)dstI[e]; }
    else           { s = e - Ereal;    d = s; }            // self loop
#pragma unroll
    for (int hh = 0; hh < H; ++hh) {
        float v = a_src[(size_t)s * H + hh] + a_dst[(size_t)d * H + hh];
        v = (v > 0.0f) ? v : v * NEG_SLOPE;
        ebuf[(size_t)e * H + hh] = v;
        atomicMax(&mmax[(size_t)d * H + hh], enc_f32(v));
    }
}

// ---------------- edge pass 2: w = exp(e - m[d]); segment sum
template <int H>
__global__ void edge_expsum_kernel(const long long* __restrict__ srcI,
                                   const long long* __restrict__ dstI,
                                   float* __restrict__ ebuf,
                                   const unsigned* __restrict__ mmax,
                                   float* __restrict__ denom,
                                   int Ereal, int nnodes) {
    int e = blockIdx.x * blockDim.x + threadIdx.x;
    if (e >= Ereal + nnodes) return;
    int d = (e < Ereal) ? (int)dstI[e] : (e - Ereal);
#pragma unroll
    for (int hh = 0; hh < H; ++hh) {
        float m = dec_f32(mmax[(size_t)d * H + hh]);
        float w = __expf(ebuf[(size_t)e * H + hh] - m);
        ebuf[(size_t)e * H + hh] = w;
        atomicAdd(&denom[(size_t)d * H + hh], w);
    }
}

// ---------------- edge pass 3: out[d] += h[s] * alpha ; one wave per edge
template <int H, int DCOLS>
__global__ void edge_aggregate_kernel(const long long* __restrict__ srcI,
                                      const long long* __restrict__ dstI,
                                      const float* __restrict__ Hm,
                                      const float* __restrict__ wbuf,
                                      const float* __restrict__ denom,
                                      float* __restrict__ out,
                                      int Ereal, int nnodes) {
    const int wave = threadIdx.x >> 5;
    const int lane = threadIdx.x & 31;
    const int e = blockIdx.x * (blockDim.x >> 5) + wave;
    if (e >= Ereal + nnodes) return;
    int s, d;
    if (e < Ereal) { s = (int)srcI[e]; d = (int)dstI[e]; }
    else           { s = e - Ereal;    d = s; }

    const float* hp = Hm + (size_t)s * DCOLS;
    float* op = out + (size_t)d * DCOLS;
    __builtin_prefetch(hp, 0, 1);   // global_prefetch_b8

    float alpha[H];
#pragma unroll
    for (int hh = 0; hh < H; ++hh)
        alpha[hh] = wbuf[(size_t)e * H + hh] / (denom[(size_t)d * H + hh] + 1e-16f);

#pragma unroll
    for (int i = 0; i < DCOLS / 32; ++i) {
        int c = lane + 32 * i;                 // coalesced
        atomicAdd(&op[c], hp[c] * alpha[c / (DCOLS / H)]);
    }
}

// ---------------- y = elu(x + bias)  (elementwise, in-place capable)
__global__ void bias_elu_kernel(const float* __restrict__ in,
                                const float* __restrict__ bias,
                                float* __restrict__ out,
                                long long total, int F) {
    long long i = (long long)blockIdx.x * blockDim.x + threadIdx.x;
    if (i >= total) return;
    float v = in[i] + bias[(int)(i % F)];
    out[i] = (v > 0.0f) ? v : (__expf(v) - 1.0f);
}

// ---------------- out[n] = sum_d elu(agg2[n,d]+b2[d]) * lin_w[d] + lin_b
__global__ void final_kernel(const float* __restrict__ agg2,
                             const float* __restrict__ b2,
                             const float* __restrict__ lin_w,
                             const float* __restrict__ lin_b,
                             float* __restrict__ out, int n) {
    const int wave = threadIdx.x >> 5;
    const int lane = threadIdx.x & 31;
    const int node = blockIdx.x * (blockDim.x >> 5) + wave;
    if (node >= n) return;
    float s = 0.0f;
#pragma unroll
    for (int i = 0; i < 2; ++i) {
        int c = lane + 32 * i;
        float v = agg2[(size_t)node * 64 + c] + b2[c];
        v = (v > 0.0f) ? v : (__expf(v) - 1.0f);
        s += v * lin_w[c];
    }
#pragma unroll
    for (int m = 16; m >= 1; m >>= 1) s += __shfl_xor(s, m, 32);
    if (lane == 0) out[node] = s + lin_b[0];
}

extern "C" void kernel_launch(void* const* d_in, const int* in_sizes, int n_in,
                              void* d_out, int out_size, void* d_ws, size_t ws_size,
                              hipStream_t stream) {
    const float*     x        = (const float*)d_in[0];
    const long long* eidx     = (const long long*)d_in[1];
    const float*     W1       = (const float*)d_in[2];
    const float*     att_src1 = (const float*)d_in[3];
    const float*     att_dst1 = (const float*)d_in[4];
    const float*     b1       = (const float*)d_in[5];
    const float*     W2       = (const float*)d_in[6];
    const float*     att_src2 = (const float*)d_in[7];
    const float*     att_dst2 = (const float*)d_in[8];
    const float*     b2       = (const float*)d_in[9];
    const float*     lin_w    = (const float*)d_in[10];
    const float*     lin_b    = (const float*)d_in[11];
    float*           out      = (float*)d_out;

    const int N   = in_sizes[0] / 128;   // nodes
    const int E   = in_sizes[1] / 2;     // edges (pre self-loop)
    const int TOT = E + N;
    const long long* srcI = eidx;
    const long long* dstI = eidx + E;

    // ---- workspace layout (f32 unless noted) ----
    float* h1    = (float*)d_ws;                          // N*128
    float* agg1  = h1   + (size_t)N * 128;                // N*128 (-> z1 in place)
    float* ebuf  = agg1 + (size_t)N * 128;                // (E+N)*2
    float* asrc  = ebuf + (size_t)TOT * 2;                // N*2
    float* adst  = asrc + (size_t)N * 2;                  // N*2
    unsigned* mE = (unsigned*)(adst + (size_t)N * 2);     // N*2 (encoded max)
    float* den   = (float*)(mE + (size_t)N * 2);          // N*2
    // layer-2 aliases (h1 dead after aggregation 1)
    float* h2    = h1;                                    // N*64
    float* agg2  = h1 + (size_t)N * 64;                   // N*64

    const int THR = 256;
    const int gemmBlocks = (N + 127) / 128;
    const int nodeWaves  = (N + 7) / 8;       // 8 waves/block
    const int edgeThr    = (TOT + THR - 1) / THR;
    const int edgeWaves  = (TOT + 7) / 8;

    // ======== layer 1 ========
    hipMemsetAsync(agg1, 0, (size_t)N * 128 * sizeof(float), stream);
    wmma_gemm_kernel<128, 128><<<gemmBlocks, THR, 0, stream>>>(x, W1, h1, N);
    att_coef_kernel<<<nodeWaves, THR, 0, stream>>>(h1, att_src1, att_dst1, asrc, adst, N, 2);
    init_softmax_kernel<<<(N * 2 + THR - 1) / THR, THR, 0, stream>>>(mE, den, N * 2);
    edge_max_kernel<2><<<edgeThr, THR, 0, stream>>>(srcI, dstI, asrc, adst, ebuf, mE, E, N);
    edge_expsum_kernel<2><<<edgeThr, THR, 0, stream>>>(srcI, dstI, ebuf, mE, den, E, N);
    edge_aggregate_kernel<2, 128><<<edgeWaves, THR, 0, stream>>>(srcI, dstI, h1, ebuf, den, agg1, E, N);
    bias_elu_kernel<<<(int)(((long long)N * 128 + THR - 1) / THR), THR, 0, stream>>>(
        agg1, b1, agg1, (long long)N * 128, 128);

    // ======== layer 2 ========
    wmma_gemm_kernel<128, 64><<<gemmBlocks, THR, 0, stream>>>(agg1, W2, h2, N);
    hipMemsetAsync(agg2, 0, (size_t)N * 64 * sizeof(float), stream);
    att_coef_kernel<<<nodeWaves, THR, 0, stream>>>(h2, att_src2, att_dst2, asrc, adst, N, 1);
    init_softmax_kernel<<<(N + THR - 1) / THR, THR, 0, stream>>>(mE, den, N);
    edge_max_kernel<1><<<edgeThr, THR, 0, stream>>>(srcI, dstI, asrc, adst, ebuf, mE, E, N);
    edge_expsum_kernel<1><<<edgeThr, THR, 0, stream>>>(srcI, dstI, ebuf, mE, den, E, N);
    edge_aggregate_kernel<1, 64><<<edgeWaves, THR, 0, stream>>>(srcI, dstI, h2, ebuf, den, agg2, E, N);

    // ======== readout ========
    final_kernel<<<nodeWaves, THR, 0, stream>>>(agg2, b2, lin_w, lin_b, out, N);
}
